// TopKRouter_64733747085703
// MI455X (gfx1250) — compile-verified, matrix operand order checked by reading, not run on hardware
//
#include <hip/hip_runtime.h>
#include <hip/hip_bf16.h>

typedef __attribute__((ext_vector_type(16))) __bf16 v16bf;
typedef __attribute__((ext_vector_type(8)))  __bf16 v8bf;
typedef __attribute__((ext_vector_type(8)))  float  v8f;

#define Bsz 4
#define Ssz 4096
#define Dsz 2048
#define Nex 64
#define Ktot 8
#define Kshared 2
#define Ng (Nex - Kshared)   // 62 routed experts
#define Kg (Ktot - Kshared)  // 6 routed picks
#define Msz (Bsz * Ssz)      // 16384 tokens
#define LDS_STRIDE (Nex + 1) // pad to avoid 64-bank conflicts

// ---------------------------------------------------------------------------
// Prep: transpose Wg (D x N, bf16) -> WgT (N x D, bf16) so B-fragment loads
// are contiguous per lane (same packing as the A 16x32 layout).
// ---------------------------------------------------------------------------
__global__ void transpose_wg_kernel(const __bf16* __restrict__ wg,
                                    __bf16* __restrict__ wgt) {
    int idx = blockIdx.x * blockDim.x + threadIdx.x;  // over N*D
    if (idx >= Nex * Dsz) return;
    int n = idx / Dsz;
    int d = idx - n * Dsz;
    wgt[idx] = wg[d * Nex + n];
}

// ---------------------------------------------------------------------------
// Main: per block -> 16 tokens. 4 waves, wave w computes experts [16w,16w+16)
// via v_wmma_f32_16x16x32_bf16 over K=2048, then per-token routing from LDS.
// Also streams the block's x tile to out_x (pass-through output).
// ---------------------------------------------------------------------------
__global__ __launch_bounds__(128)
void router_kernel(const __bf16* __restrict__ x,       // M x D
                   const __bf16* __restrict__ wgt,     // N x D (transposed)
                   const float*  __restrict__ biases,  // N
                   __bf16* __restrict__ out_x,         // M x D
                   int*    __restrict__ out_mask,      // N x M
                   float*  __restrict__ out_s)         // N x M
{
    __shared__ float scores[16 * LDS_STRIDE];
    __shared__ float bsh[Ng];

    const int tid  = threadIdx.x;
    const int wave = tid >> 5;
    const int lane = tid & 31;
    const int m0   = blockIdx.x * 16;   // token base
    const int n0   = wave * 16;         // expert base for this wave

    if (tid < Ng) bsh[tid] = biases[tid];

    // Fragment addressing per CDNA5 16-bit A 16x32 layout:
    // lanes 0-15 (row = lane) hold K 0..7 (v0-3) and 16..23 (v4-7);
    // lanes 16-31 hold K 8..15 and 24..31.  Two contiguous 16B loads/lane.
    const int rowA  = m0 + (lane & 15);
    const int rowB  = n0 + (lane & 15);
    const int kHalf = (lane >> 4) * 8;

    const __bf16* aBase = x   + (size_t)rowA * Dsz + kHalf;
    const __bf16* bBase = wgt + (size_t)rowB * Dsz + kHalf;

    v8f acc = {};
    for (int kk = 0; kk < Dsz; kk += 32) {
        v8bf alo = *(const v8bf*)(aBase + kk);
        v8bf ahi = *(const v8bf*)(aBase + kk + 16);
        v8bf blo = *(const v8bf*)(bBase + kk);
        v8bf bhi = *(const v8bf*)(bBase + kk + 16);
        v16bf a, b;
#pragma unroll
        for (int i = 0; i < 8; ++i) {
            a[i] = alo[i]; a[i + 8] = ahi[i];
            b[i] = blo[i]; b[i + 8] = bhi[i];
        }
        acc = __builtin_amdgcn_wmma_f32_16x16x32_bf16(
            /*neg_a=*/false, a, /*neg_b=*/false, b,
            /*c_mod=*/(short)0, acc, /*reuse_a=*/false, /*reuse_b=*/false);
    }

    // C/D layout: lane%16 = N column, VGPR r = row r (lanes 0-15) / r+8 (16-31)
    {
        const int n  = n0 + (lane & 15);
        const int hi = (lane >> 4) * 8;
#pragma unroll
        for (int r = 0; r < 8; ++r)
            scores[(r + hi) * LDS_STRIDE + n] = acc[r];
    }
    __syncthreads();

    // Pass-through copy of x tile: 16 rows * 2048 bf16 = 4096 uint4
    {
        const uint4* src = (const uint4*)(x     + (size_t)m0 * Dsz);
        uint4*       dst = (uint4*)      (out_x + (size_t)m0 * Dsz);
#pragma unroll
        for (int i = 0; i < 32; ++i)
            dst[tid + i * 128] = src[tid + i * 128];
    }

    // Per-token routing: one thread per token (threads 0..15).
    if (tid < 16) {
        const int m  = m0 + tid;
        float*    sc = &scores[tid * LDS_STRIDE];

        // routed-group softmax over experts [0, Ng)
        float mx = -3.0e38f;
        for (int n = 0; n < Ng; ++n) mx = fmaxf(mx, sc[n]);
        float sum = 0.f;
        for (int n = 0; n < Ng; ++n) {
            float e = __expf(sc[n] - mx);
            sc[n] = e;           // stage un-normalized probs back in LDS
            sum += e;
        }
        const float inv = 1.f / sum;

        // biased top-Kg selection (bias affects selection only)
        unsigned long long sel = 0ull;
        float denom = 0.f;
        for (int k = 0; k < Kg; ++k) {
            int   best  = 0;
            float bestv = -3.0e38f;
            for (int n = 0; n < Ng; ++n) {
                float v = sc[n] * inv + bsh[n];
                bool  taken = (sel >> n) & 1ull;
                if (!taken && v > bestv) { bestv = v; best = n; }
            }
            sel |= 1ull << best;
            denom += sc[best] * inv;
        }
        const float invd = 1.f / denom;

        for (int n = 0; n < Ng; ++n) {
            const bool taken = (sel >> n) & 1ull;
            out_mask[(size_t)n * Msz + m] = taken ? 1 : 0;
            out_s[(size_t)n * Msz + m]    = taken ? sc[n] * inv * invd : 0.f;
        }

        // shared-group softmax over the last 2 experts (denominator = 1)
        const float s0  = sc[Ng], s1 = sc[Ng + 1];
        const float mx2 = fmaxf(s0, s1);
        const float e0  = __expf(s0 - mx2);
        const float e1  = __expf(s1 - mx2);
        const float is  = 1.f / (e0 + e1);
        out_mask[(size_t)Ng * Msz + m]       = 1;
        out_mask[(size_t)(Ng + 1) * Msz + m] = 1;
        out_s[(size_t)Ng * Msz + m]          = e0 * is;
        out_s[(size_t)(Ng + 1) * Msz + m]    = e1 * is;
    }
}

extern "C" void kernel_launch(void* const* d_in, const int* in_sizes, int n_in,
                              void* d_out, int out_size, void* d_ws, size_t ws_size,
                              hipStream_t stream) {
    (void)in_sizes; (void)n_in; (void)out_size; (void)ws_size;

    const __bf16* x      = (const __bf16*)d_in[0];   // B*S*D bf16
    const __bf16* wg     = (const __bf16*)d_in[1];   // D*N  bf16
    const float*  biases = (const float*) d_in[2];   // N    fp32

    char* outc = (char*)d_out;
    __bf16* out_x    = (__bf16*)outc;                                    // M*D bf16
    int*    out_mask = (int*)  (outc + (size_t)Msz * Dsz * 2);           // N*M int32
    float*  out_s    = (float*)(outc + (size_t)Msz * Dsz * 2
                                     + (size_t)Nex * Msz * 4);           // N*M fp32

    __bf16* wgt = (__bf16*)d_ws;                     // N*D bf16 scratch (256 KB)

    transpose_wg_kernel<<<(Nex * Dsz + 255) / 256, 256, 0, stream>>>(wg, wgt);
    router_kernel<<<Msz / 16, 128, 0, stream>>>(x, wgt, biases,
                                                out_x, out_mask, out_s);
}
